// INT4Linear_21182778704936
// MI455X (gfx1250) — compile-verified, matrix operand order checked by reading, not run on hardware
//
#include <hip/hip_runtime.h>
#include <hip/hip_bf16.h>

typedef __attribute__((ext_vector_type(16))) _Float16 v16h;
typedef __attribute__((ext_vector_type(8)))  _Float16 v8h;
typedef __attribute__((ext_vector_type(2)))  _Float16 h2;
typedef __attribute__((ext_vector_type(8)))  float    v8f;
typedef __attribute__((ext_vector_type(4)))  float    v4f;
typedef __attribute__((ext_vector_type(4)))  int      v4i;
typedef __attribute__((ext_vector_type(8)))  int      v8i;
typedef __attribute__((ext_vector_type(4)))  unsigned v4u;

// Problem constants (from reference)
#define DIM_M 256      // B*S = 2*128
#define DIM_N 12288    // OUT
#define DIM_K 4096     // IN
#define GROUP 128      // quant group size along K

// Tiling
#define KC    64       // K chunk (2 chunks per quant group)
#define TM    128      // M per block (8 wmma M-tiles -> B frag reused 8x)
#define TN    128      // N per block (8 waves x 16)
#define LDSK  (KC + 8) // padded row stride in halves: 144B rows, bank-rotated
#define NCHUNK (DIM_K / KC)

// ---------------- pre-pass: x fp32 -> f16 into workspace ----------------
__global__ __launch_bounds__(256)
void x_to_f16_kernel(const float* __restrict__ x, _Float16* __restrict__ xh)
{
    const int i = (blockIdx.x * 256 + threadIdx.x) * 8;  // DIM_M*DIM_K = 1048576 elems
    v4f f0 = *(const v4f*)(x + i);
    v4f f1 = *(const v4f*)(x + i + 4);
    v8h h;
    h[0] = (_Float16)f0[0]; h[1] = (_Float16)f0[1];
    h[2] = (_Float16)f0[2]; h[3] = (_Float16)f0[3];
    h[4] = (_Float16)f1[0]; h[5] = (_Float16)f1[1];
    h[6] = (_Float16)f1[2]; h[7] = (_Float16)f1[3];
    *(v8h*)(xh + i) = h;
}

// ---------------- A-tile staging: TDM if available, else async global->LDS ----------------
#if __has_builtin(__builtin_amdgcn_tensor_load_to_lds)
#define USE_TDM 1
#else
#define USE_TDM 0
#endif

#if USE_TDM
// One wave issues a single TENSOR_LOAD_TO_LDS for the whole 128x64 f16 tile.
// D# bit layout per CDNA5 ISA ch.8: group0 = {flags, lds_addr, global_addr, type},
// group1 = {pad cfg, tensor dims, tile dims, strides}. HW pads 4 DWORDs every
// 32 DWORDs (one 64-half row) -> LDSK = 72-half row stride, bank-rotated.
__device__ __forceinline__ void tdm_issue_tile(const _Float16* xh, unsigned ldsBase,
                                               int mBase, int k0)
{
    const unsigned long long ga =
        (unsigned long long)(uintptr_t)xh +
        ((unsigned long long)mBase * DIM_K + (unsigned)k0) * 2ull;

    v4u g0;
    g0[0] = 1u;                                   // count=1, load-descriptor, no gather
    g0[1] = ldsBase;                              // lds_addr (bytes)
    g0[2] = (unsigned)ga;                         // global_addr[31:0]
    g0[3] = (unsigned)((ga >> 32) & 0x01FFFFFFu)  // global_addr[56:32]
          | (2u << 30);                           // type = 2 ("image")

    v8i g1;
    g1[0] = (int)((1u << 16)      // data_size = 1 -> 2 bytes
                | (1u << 20)      // pad_enable
                | (4u << 22)      // pad_interval: 2^(4+1) = 32 DWORDs (= one 64-half row)
                | (3u << 25));    // pad_amount: 3+1 = 4 DWORDs (= 8 halves)
    g1[1] = (int)(((unsigned)DIM_K & 0xFFFFu) << 16);   // tensor_dim0[15:0] = 4096
    g1[2] = (int)(((unsigned)DIM_M & 0xFFFFu) << 16);   // tensor_dim0[31:16]=0 | tensor_dim1[15:0]=256
    g1[3] = (int)(((unsigned)KC & 0xFFFFu) << 16);      // tensor_dim1[31:16]=0 | tile_dim0 = 64
    g1[4] = (int)((unsigned)TM & 0xFFFFu);              // tile_dim1 = 128 | tile_dim2 = 0
    g1[5] = (int)DIM_K;                                 // tensor_dim0_stride[31:0] = 4096
    g1[6] = 0;                                          // stride0[47:32]=0 | dim1_stride[15:0]=0
    g1[7] = 0;

    const v4i z4 = {0, 0, 0, 0};
#if defined(__clang_major__) && __clang_major__ >= 23
    const v8i z8 = {0, 0, 0, 0, 0, 0, 0, 0};
    __builtin_amdgcn_tensor_load_to_lds(g0, g1, z4, z4, z8, 0);
#else
    __builtin_amdgcn_tensor_load_to_lds(g0, g1, z4, z4, 0);
#endif
}
#endif

// ---------------- main GEMM ----------------
__global__ __launch_bounds__(256)
void int4_linear_wmma_kernel(const _Float16* __restrict__ xh,     // (M, K) f16
                             const int*      __restrict__ pw,     // (N, K/2) one byte per int32
                             const float*    __restrict__ scales, // (N, K/GROUP)
                             float*          __restrict__ out)    // (M, N)
{
    __shared__ __align__(16) _Float16 As[2][TM * LDSK];   // ~36 KB double-buffered

    const int t    = threadIdx.x;
    const int wave = t >> 5;          // 0..7
    const int lane = t & 31;
    const int l16  = lane & 15;
    const int hi   = (lane >> 4) & 1; // 0: K 0-7/16-23 ; 1: K 8-15/24-31

    const int nBase = blockIdx.x * TN;
    const int mBase = blockIdx.y * TM;
    const int n     = nBase + wave * 16 + l16;          // this lane's output column
    const long long pwRow = (long long)n * (DIM_K / 2); // 2048 int32 per row

    const unsigned ldsBase0 = (unsigned)(uintptr_t)&As[0][0];
    const unsigned ldsBase1 = (unsigned)(uintptr_t)&As[1][0];

    v8f acc[8];
#pragma unroll
    for (int i = 0; i < 8; ++i) acc[i] = (v8f)0.0f;

    // ---- stage issue/wait helpers (TDM or per-thread async fallback) ----
#if USE_TDM
#define STAGE_ISSUE(cc)                                                     \
    do { if (wave == 0)                                                     \
        tdm_issue_tile(xh, ((cc) & 1) ? ldsBase1 : ldsBase0, mBase, (cc) * KC); } while (0)
#if __has_builtin(__builtin_amdgcn_s_wait_tensorcnt)
#define STAGE_WAIT() do { if (wave == 0) __builtin_amdgcn_s_wait_tensorcnt(0); } while (0)
#else
#define STAGE_WAIT() do { if (wave == 0) asm volatile("s_wait_tensorcnt 0x0" ::: "memory"); } while (0)
#endif
#else
#define STAGE_ISSUE(cc)                                                          \
    do {                                                                         \
        const int _k0 = (cc) * KC;                                               \
        const unsigned _lb = ((cc) & 1) ? ldsBase1 : ldsBase0;                   \
        _Pragma("unroll")                                                        \
        for (int _r = 0; _r < 4; ++_r) {                                         \
            const int _idx = _r * 2048 + t * 8;   /* halves; TM*KC = 8192 */     \
            const int _row = _idx >> 6;           /* /KC */                      \
            const int _col = _idx & (KC - 1);                                    \
            const unsigned _go = (unsigned)(((mBase + _row) * DIM_K + _k0 + _col) * 2); \
            const unsigned _la = _lb + (unsigned)((_row * LDSK + _col) * 2);     \
            asm volatile("global_load_async_to_lds_b128 %0, %1, %2 offset:0"     \
                         :: "v"(_la), "v"(_go), "s"(xh) : "memory");             \
        }                                                                        \
    } while (0)
#if __has_builtin(__builtin_amdgcn_s_wait_asynccnt)
#define STAGE_WAIT() __builtin_amdgcn_s_wait_asynccnt(0)
#else
#define STAGE_WAIT() asm volatile("s_wait_asynccnt 0x0" ::: "memory")
#endif
#endif

    // prologue: stage chunk 0 into buffer 0
    STAGE_ISSUE(0);

    for (int c = 0; c < NCHUNK; ++c) {
        const int k0 = c * KC;

        STAGE_WAIT();        // chunk c resident in LDS (issuer side)
        __syncthreads();     // publish chunk c; all waves done reading chunk c-1
        if (c + 1 < NCHUNK)
            STAGE_ISSUE(c + 1);   // overlap next copy with this chunk's math

        const _Float16* Ab = ((c & 1) ? As[1] : As[0]);

        // per-lane dequant constants (quant group = 128 = 2 chunks):
        //   (nib - 8) * s  ==  f16(0x6400|nib) * s + (-1032 * s)
        const float s = scales[(long long)n * (DIM_K / GROUP) + (k0 >> 7)];
        const _Float16 sh = (_Float16)s;
        const _Float16 ch = (_Float16)(-1032.0f * s);
        const h2 s2 = {sh, sh};
        const h2 c2 = {ch, ch};

        // prefetch next weight chunk (emits global_prefetch_b8)
        if (k0 + KC < DIM_K)
            __builtin_prefetch(pw + pwRow + ((k0 + KC) >> 1), 0, 1);

#pragma unroll
        for (int kk = 0; kk < KC; kk += 32) {
            // ---- build B fragment: W[n, k0+kk .. +32] dequantized to f16 ----
            const int kSel = kk + (hi ? 8 : 0);
            const int* p0 = pw + pwRow + ((k0 + kSel) >> 1);        // K kSel..kSel+7
            const int* p1 = pw + pwRow + ((k0 + kSel + 16) >> 1);   // K kSel+16..kSel+23
            v4i e0 = *(const v4i*)p0;
            v4i e1 = *(const v4i*)p1;
            v8i bbits;
#pragma unroll
            for (int j = 0; j < 4; ++j) {
                const unsigned u0 = (unsigned)e0[j];
                const unsigned b0 = 0x64006400u | (u0 & 0xFu) | ((u0 << 12) & 0x000F0000u);
                h2 w0 = __builtin_bit_cast(h2, b0) * s2 + c2;
                bbits[j] = __builtin_bit_cast(int, w0);

                const unsigned u1 = (unsigned)e1[j];
                const unsigned b1 = 0x64006400u | (u1 & 0xFu) | ((u1 << 12) & 0x000F0000u);
                h2 w1 = __builtin_bit_cast(h2, b1) * s2 + c2;
                bbits[4 + j] = __builtin_bit_cast(int, w1);
            }
            const v16h b = __builtin_bit_cast(v16h, bbits);

            // ---- 8 M-tiles reuse this B fragment ----
#pragma unroll
            for (int mt = 0; mt < 8; ++mt) {
                const _Float16* arow = &Ab[(mt * 16 + l16) * LDSK + kSel];
                v8h a0 = *(const v8h*)(arow);       // K kSel..+7
                v8h a1 = *(const v8h*)(arow + 16);  // K kSel+16..+23
                v16h a;
#pragma unroll
                for (int j = 0; j < 8; ++j) { a[j] = a0[j]; a[8 + j] = a1[j]; }
                acc[mt] = __builtin_amdgcn_wmma_f32_16x16x32_f16(
                    /*neg_a=*/false, a, /*neg_b=*/false, b,
                    /*c_mod=*/(short)0, acc[mt],
                    /*reuse_a=*/false, /*reuse_b=*/false);
            }
        }
    }

    // ---- write C: VGPR r -> M = r + 8*hi, N = lane%16 ----
#pragma unroll
    for (int mt = 0; mt < 8; ++mt) {
#pragma unroll
        for (int r = 0; r < 8; ++r) {
            const int mg = mBase + mt * 16 + r + hi * 8;
            out[(long long)mg * DIM_N + n] = acc[mt][r];
        }
    }
}

extern "C" void kernel_launch(void* const* d_in, const int* in_sizes, int n_in,
                              void* d_out, int out_size, void* d_ws, size_t ws_size,
                              hipStream_t stream) {
    const float* x      = (const float*)d_in[0];
    const int*   pw     = (const int*)d_in[1];
    const float* scales = (const float*)d_in[2];
    float*       out    = (float*)d_out;
    _Float16*    xh     = (_Float16*)d_ws;   // DIM_M*DIM_K*2 = 2 MB scratch
    (void)in_sizes; (void)n_in; (void)out_size; (void)ws_size;

    // pass 1: convert activations fp32 -> f16 once (1048576 elems, 8 per thread)
    x_to_f16_kernel<<<dim3((DIM_M * DIM_K) / (256 * 8)), dim3(256), 0, stream>>>(x, xh);

    // pass 2: WMMA GEMM with TDM/async double-buffered A staging + int4 dequant
    dim3 grid(DIM_N / TN, DIM_M / TM);  // (96, 2)
    dim3 block(256);                    // 8 wave32
    int4_linear_wmma_kernel<<<grid, block, 0, stream>>>(xh, pw, scales, out);
}
